// BiSE3Transformer_1116691497437
// MI455X (gfx1250) — matrix operand order repaired and checked by reading, not compiled, not run on hardware
//
#include <hip/hip_runtime.h>
#include <hip/hip_bf16.h>

#define NN   50000
#define EE   800000
#define CC   32
#define HRADK 32
#define LLAY 4
#define L1   (LLAY + 1)
#define NHEADS 4
#define NRADC (7 * CC)
#define NTW2  (NRADC / 16)   // 14 column tiles in W2
#define EPSF 1e-6f

typedef _Float16 v16h __attribute__((ext_vector_type(16)));
typedef float    v8f  __attribute__((ext_vector_type(8)));

// ---------------------------------------------------------------------------
// helpers
// ---------------------------------------------------------------------------
__device__ __forceinline__ float atomicMaxF(float* addr, float val) {
  unsigned int* ua = (unsigned int*)addr;
  unsigned int old = *ua;
  while (true) {
    float f = __uint_as_float(old);
    if (f >= val) break;
    unsigned int assumed = old;
    old = atomicCAS(ua, assumed, __float_as_uint(val));
    if (old == assumed) break;
  }
  return __uint_as_float(old);
}

__device__ __forceinline__ v8f wmma32(v16h a, v16h b, v8f c) {
  // D = A(16x32 f16) x B(32x16 f16) + C(16x16 f32)
  return __builtin_amdgcn_wmma_f32_16x16x32_f16(
      /*neg_a=*/false, a, /*neg_b=*/false, b,
      /*c_mod=*/(short)0, c, /*reuse_a=*/false, /*reuse_b=*/false);
}

// A-operand from f32 matrix with contiguous channels (elem stride 1).
// Lanes 0-15 need row floats [koff, koff+8) and [koff+16, koff+24), koff=0;
// lanes 16-31 the same with koff=8. All chunks 32B aligned -> float4 loads.
__device__ __forceinline__ v16h load_a16_c(const float* X, int rowBase,
                                           int rowStride, int lane) {
  int m    = lane & 15;
  int koff = (lane & 16) ? 8 : 0;
  const float* row = X + (size_t)(rowBase + m) * rowStride + koff;
  float4 p0 = ((const float4*)row)[0];
  float4 p1 = ((const float4*)row)[1];
  float4 p2 = ((const float4*)(row + 16))[0];
  float4 p3 = ((const float4*)(row + 16))[1];
  v16h a;
  a[0]  = (_Float16)p0.x; a[1]  = (_Float16)p0.y; a[2]  = (_Float16)p0.z; a[3]  = (_Float16)p0.w;
  a[4]  = (_Float16)p1.x; a[5]  = (_Float16)p1.y; a[6]  = (_Float16)p1.z; a[7]  = (_Float16)p1.w;
  a[8]  = (_Float16)p2.x; a[9]  = (_Float16)p2.y; a[10] = (_Float16)p2.z; a[11] = (_Float16)p2.w;
  a[12] = (_Float16)p3.x; a[13] = (_Float16)p3.y; a[14] = (_Float16)p3.z; a[15] = (_Float16)p3.w;
  return a;
}

// A-operand from strided f32 matrix (for f1/o1 xyz components, elem stride 3).
__device__ __forceinline__ v16h load_a16_s(const float* X, int rowBase,
                                           int rowStride, int elemStride, int lane) {
  v16h a;
  int m    = lane & 15;
  int koff = (lane & 16) ? 8 : 0;
  const float* row = X + (size_t)(rowBase + m) * rowStride;
#pragma unroll
  for (int j = 0; j < 8; ++j) {
    int k = ((j < 4) ? (2 * j) : (2 * j + 8)) + koff;
    a[2 * j]     = (_Float16)row[k * elemStride];
    a[2 * j + 1] = (_Float16)row[(k + 1) * elemStride];
  }
  return a;
}

// A from radial MLP hidden layer: h_k = relu(rlen*W1[k]+b1[k])
__device__ __forceinline__ v16h build_a_radial(float rl, const float* W1l,
                                               const float* b1l, int lane) {
  v16h a;
  int koff = (lane & 16) ? 8 : 0;
#pragma unroll
  for (int j = 0; j < 8; ++j) {
    int k = ((j < 4) ? (2 * j) : (2 * j + 8)) + koff;
    float h0 = fmaxf(rl * W1l[k]     + b1l[k],     0.f);
    float h1 = fmaxf(rl * W1l[k + 1] + b1l[k + 1], 0.f);
    a[2 * j]     = (_Float16)h0;
    a[2 * j + 1] = (_Float16)h1;
  }
  return a;
}

// Pre-swizzled B-operand: [tile][lane][16 f16] contiguous -> 2x b128 per lane.
__device__ __forceinline__ v16h load_bp(const _Float16* Wpl, int nb, int lane) {
  return *(const v16h*)(Wpl + ((size_t)nb * 32 + lane) * 16);
}

// ---------------------------------------------------------------------------
// setup kernels
// ---------------------------------------------------------------------------
// Pack one 32x16 B tile into per-lane WMMA layout:
// out[lane][j] = W[(j + (j<8?0:8) + (lane&16?8:0)) * ncols + tile*16 + (lane&15)]
__device__ __forceinline__ void pack_tile(const float* Wl, int ncols, int tile,
                                          int lane, _Float16* out16) {
  int n    = tile * 16 + (lane & 15);
  int koff = (lane & 16) ? 8 : 0;
#pragma unroll
  for (int j = 0; j < 16; ++j) out16[j] = (_Float16)Wl[(j + ((j < 8) ? 0 : 8) + koff) * ncols + n];
}

__global__ void k_pack_b(const float* W2, const float* Wq, const float* Wo0,
                         const float* Wo1, const float* Ws0, const float* Ws1,
                         _Float16* W2p, _Float16* Wqp, _Float16* Wo0p,
                         _Float16* Wo1p, _Float16* Ws0p, _Float16* Ws1p) {
  int i = blockIdx.x * blockDim.x + threadIdx.x;  // flattened (layer, tile, lane)
  if (i < L1 * NTW2 * 32) {
    int lane = i & 31;
    int tile = (i >> 5) % NTW2;
    int l    = (i >> 5) / NTW2;
    pack_tile(W2 + (size_t)l * HRADK * NRADC, NRADC, tile, lane,
              W2p + (size_t)i * 16);
  }
  if (i < L1 * 2 * 32) {
    int lane = i & 31;
    int tile = (i >> 5) & 1;
    int l    = (i >> 5) >> 1;
    size_t woff = (size_t)l * CC * CC;
    size_t ooff = (size_t)i * 16;
    pack_tile(Wq  + woff, CC, tile, lane, Wqp  + ooff);
    pack_tile(Wo0 + woff, CC, tile, lane, Wo0p + ooff);
    pack_tile(Wo1 + woff, CC, tile, lane, Wo1p + ooff);
    pack_tile(Ws0 + woff, CC, tile, lane, Ws0p + ooff);
    pack_tile(Ws1 + woff, CC, tile, lane, Ws1p + ooff);
  }
}

__global__ void k_edge_geom(const float* pos, const int* src, const int* dst,
                            float* rlen, float* u) {
  int e = blockIdx.x * blockDim.x + threadIdx.x;
  if (e >= EE) return;
  int s = src[e], d = dst[e];
  float rx = pos[d * 3 + 0] - pos[s * 3 + 0];
  float ry = pos[d * 3 + 1] - pos[s * 3 + 1];
  float rz = pos[d * 3 + 2] - pos[s * 3 + 2];
  float rl = sqrtf(rx * rx + ry * ry + rz * rz);
  rlen[e]  = rl;
  float inv = 1.f / (rl + EPSF);
  u[e * 3 + 0] = rx * inv;
  u[e * 3 + 1] = ry * inv;
  u[e * 3 + 2] = rz * inv;
}

__global__ void k_zero_acc(float* mbuf, float* zbuf, float* o0, float* o1) {
  int i = blockIdx.x * blockDim.x + threadIdx.x;  // over N*C
  if (i >= NN * CC) return;
  o0[i] = 0.f;
  o1[(size_t)i * 3 + 0] = 0.f;
  o1[(size_t)i * 3 + 1] = 0.f;
  o1[(size_t)i * 3 + 2] = 0.f;
  int c = i & (CC - 1);
  if (c < NHEADS) {
    int n = i >> 5;
    mbuf[n * NHEADS + c] = -1e30f;
    zbuf[n * NHEADS + c] = 0.f;
  }
}

// ---------------------------------------------------------------------------
// q = f0 @ Wq   (WMMA, K=32)
// ---------------------------------------------------------------------------
__global__ void k_node_q(const float* f0, const _Float16* Wqpl, float* q) {
  int wave = (blockIdx.x * blockDim.x + threadIdx.x) >> 5;
  int lane = threadIdx.x & 31;
  if (wave >= NN / 16) return;  // wave-uniform guard: EXEC stays all-ones
  int base  = wave * 16;
  v16h a    = load_a16_c(f0, base, CC, lane);
  int rbase = base + ((lane & 16) ? 8 : 0);
#pragma unroll
  for (int t = 0; t < 2; ++t) {
    v16h b  = load_bp(Wqpl, t, lane);
    v8f acc = {};
    acc = wmma32(a, b, acc);
    int c = t * 16 + (lane & 15);
#pragma unroll
    for (int r = 0; r < 8; ++r) q[(rbase + r) * CC + c] = acc[r];
  }
}

// ---------------------------------------------------------------------------
// Pass 1: attention logits (radial k-slices via WMMA) + segment max
// ---------------------------------------------------------------------------
__global__ void k_edge_logits(const float* rlen, const float* u,
                              const int* src, const int* dst,
                              const float* f0, const float* f1, const float* q,
                              const float* W1l, const float* b1l,
                              const _Float16* W2pl, const float* b2l,
                              int heads, float* logits, float* mbuf) {
  int wave = (blockIdx.x * blockDim.x + threadIdx.x) >> 5;
  int lane = threadIdx.x & 31;
  if (wave >= EE / 16) return;
  int base = wave * 16;
  float rl = rlen[base + (lane & 15)];
  v16h a   = build_a_radial(rl, W1l, b1l, lane);
  int rowoff = (lane & 16) ? 8 : 0;

  // hoist per-edge gathers out of the channel-tile loop
  int eS[8], eD[8];
  float ux[8], uy[8], uz[8];
#pragma unroll
  for (int r = 0; r < 8; ++r) {
    int e = base + rowoff + r;
    eS[r] = src[e];
    eD[r] = dst[e];
    ux[r] = u[e * 3 + 0];
    uy[r] = u[e * 3 + 1];
    uz[r] = u[e * 3 + 2];
  }

  float psum[8];
#pragma unroll
  for (int r = 0; r < 8; ++r) psum[r] = 0.f;

#pragma unroll
  for (int t = 0; t < 2; ++t) {
    v16h b00 = load_bp(W2pl, 0 * 2 + t, lane);  // rk00 tile
    v16h b10 = load_bp(W2pl, 1 * 2 + t, lane);  // rk10 tile
    v8f a00 = {}; a00 = wmma32(a, b00, a00);
    v8f a10 = {}; a10 = wmma32(a, b10, a10);
    int c = t * 16 + (lane & 15);
    float bk00 = b2l[c], bk10 = b2l[CC + c];
    float p[8];
#pragma unroll
    for (int r = 0; r < 8; ++r) {
      float f0s = f0[eS[r] * CC + c];
      const float* f1r = f1 + (size_t)eS[r] * CC * 3 + c * 3;
      float dot1 = f1r[0] * ux[r] + f1r[1] * uy[r] + f1r[2] * uz[r];
      float k0 = (a00[r] + bk00) * f0s + (a10[r] + bk10) * dot1;
      p[r] = q[eD[r] * CC + c] * k0;
    }
    if (heads == NHEADS) {
#pragma unroll
      for (int r = 0; r < 8; ++r) {
        float v = p[r];
        v += __shfl_xor(v, 1, 32);
        v += __shfl_xor(v, 2, 32);
        v += __shfl_xor(v, 4, 32);
        p[r] = v * 0.35355339059327373f;  // 1/sqrt(8)
      }
      if ((lane & 7) == 0) {
        int head = t * 2 + ((lane >> 3) & 1);
#pragma unroll
        for (int r = 0; r < 8; ++r) {
          int e = base + rowoff + r;
          logits[e * NHEADS + head] = p[r];
          atomicMaxF(&mbuf[eD[r] * NHEADS + head], p[r]);
        }
      }
    } else {
#pragma unroll
      for (int r = 0; r < 8; ++r) psum[r] += p[r];
    }
  }

  if (heads == 1) {
#pragma unroll
    for (int r = 0; r < 8; ++r) {
      float v = psum[r];
      v += __shfl_xor(v, 1, 32);
      v += __shfl_xor(v, 2, 32);
      v += __shfl_xor(v, 4, 32);
      v += __shfl_xor(v, 8, 32);
      psum[r] = v * 0.17677669529663687f;  // 1/sqrt(32)
    }
    if ((lane & 15) == 0) {
#pragma unroll
      for (int r = 0; r < 8; ++r) {
        int e = base + rowoff + r;
        logits[e * NHEADS] = psum[r];
        atomicMaxF(&mbuf[eD[r] * NHEADS], psum[r]);
      }
    }
  }
}

// ---------------------------------------------------------------------------
// Pass 2: e = exp(logit - m[dst]); z = segment_sum(e)
// ---------------------------------------------------------------------------
__global__ void k_edge_exp(const float* logits, const int* dst, const float* mbuf,
                           float* eexp, float* zbuf, int heads) {
  int i = blockIdx.x * blockDim.x + threadIdx.x;
  if (i >= EE * heads) return;
  int e = i / heads;
  int h = i - e * heads;
  int d = dst[e];
  float ev = expf(logits[e * NHEADS + h] - mbuf[d * NHEADS + h]);
  eexp[e * NHEADS + h] = ev;
  atomicAdd(&zbuf[d * NHEADS + h], ev);
}

// ---------------------------------------------------------------------------
// Pass 3: values (radial v-slices via WMMA), alpha-weighted scatter into o0/o1
// ---------------------------------------------------------------------------
__global__ void k_edge_values(const float* rlen, const float* u,
                              const int* src, const int* dst,
                              const float* f0, const float* f1,
                              const float* eexp, const float* zbuf,
                              const float* W1l, const float* b1l,
                              const _Float16* W2pl, const float* b2l,
                              int heads, float* o0, float* o1) {
  int wave = (blockIdx.x * blockDim.x + threadIdx.x) >> 5;
  int lane = threadIdx.x & 31;
  if (wave >= EE / 16) return;
  int base = wave * 16;
  float rl = rlen[base + (lane & 15)];
  v16h a   = build_a_radial(rl, W1l, b1l, lane);
  int rowoff = (lane & 16) ? 8 : 0;

  int eS[8], eD[8];
  float ux[8], uy[8], uz[8];
#pragma unroll
  for (int r = 0; r < 8; ++r) {
    int e = base + rowoff + r;
    eS[r] = src[e];
    eD[r] = dst[e];
    ux[r] = u[e * 3 + 0];
    uy[r] = u[e * 3 + 1];
    uz[r] = u[e * 3 + 2];
  }

#pragma unroll
  for (int t = 0; t < 2; ++t) {
    int c = t * 16 + (lane & 15);
    v8f av00 = {}, av10 = {}, av11 = {}, av01 = {}, av11b = {};
    av00  = wmma32(a, load_bp(W2pl, 2 * 2 + t, lane), av00);
    av10  = wmma32(a, load_bp(W2pl, 3 * 2 + t, lane), av10);
    av11  = wmma32(a, load_bp(W2pl, 4 * 2 + t, lane), av11);
    av01  = wmma32(a, load_bp(W2pl, 5 * 2 + t, lane), av01);
    av11b = wmma32(a, load_bp(W2pl, 6 * 2 + t, lane), av11b);
    float bv00 = b2l[2 * CC + c], bv10 = b2l[3 * CC + c], bv11 = b2l[4 * CC + c];
    float bv01 = b2l[5 * CC + c], bv11b = b2l[6 * CC + c];
    int head = (heads == 1) ? 0 : (t * 2 + ((lane >> 3) & 1));
#pragma unroll
    for (int r = 0; r < 8; ++r) {
      int e = base + rowoff + r;
      int s = eS[r], d = eD[r];
      float alpha = eexp[e * NHEADS + head] / (zbuf[d * NHEADS + head] + EPSF);
      float f0s = f0[s * CC + c];
      const float* f1r = f1 + (size_t)s * CC * 3 + c * 3;
      float fx = f1r[0], fy = f1r[1], fz = f1r[2];
      float dot1 = fx * ux[r] + fy * uy[r] + fz * uz[r];
      float v0 = (av00[r] + bv00) * f0s + (av10[r] + bv10) * dot1;
      atomicAdd(&o0[d * CC + c], alpha * v0);
      float r11 = av11[r] + bv11;
      float sc  = (av01[r] + bv01) * f0s + (av11b[r] + bv11b) * dot1;
      float* o1p = o1 + ((size_t)d * CC + c) * 3;
      atomicAdd(&o1p[0], alpha * (r11 * fx + sc * ux[r]));
      atomicAdd(&o1p[1], alpha * (r11 * fy + sc * uy[r]));
      atomicAdd(&o1p[2], alpha * (r11 * fz + sc * uz[r]));
    }
  }
}

// ---------------------------------------------------------------------------
// Node update: out0 = o0@Wo0 + f0@Ws0 ; out1 = o1@Wo1 + f1@Ws1 ; GProjRelu
// All GEMMs chained through the WMMA C operand.
// ---------------------------------------------------------------------------
__global__ void k_node_out(const float* f0, const float* f1,
                           const float* o0, const float* o1,
                           const _Float16* Wo0pl, const _Float16* Wo1pl,
                           const _Float16* Ws0pl, const _Float16* Ws1pl,
                           const float* bgl, int doAct,
                           float* f0out, float* f1out) {
  int wave = (blockIdx.x * blockDim.x + threadIdx.x) >> 5;
  int lane = threadIdx.x & 31;
  if (wave >= NN / 16) return;
  int base = wave * 16;
  v16h aO0 = load_a16_c(o0, base, CC, lane);
  v16h aF0 = load_a16_c(f0, base, CC, lane);
  v16h aO1[3], aF1[3];
#pragma unroll
  for (int d = 0; d < 3; ++d) {
    aO1[d] = load_a16_s(o1 + d, base, CC * 3, 3, lane);
    aF1[d] = load_a16_s(f1 + d, base, CC * 3, 3, lane);
  }
  int rbase = base + ((lane & 16) ? 8 : 0);

#pragma unroll
  for (int t = 0; t < 2; ++t) {
    int c = t * 16 + (lane & 15);
    v16h bWo0 = load_bp(Wo0pl, t, lane);
    v16h bWs0 = load_bp(Ws0pl, t, lane);
    v8f acc0 = {};
    acc0 = wmma32(aF0, bWs0, acc0);
    acc0 = wmma32(aO0, bWo0, acc0);
    v16h bWo1 = load_bp(Wo1pl, t, lane);
    v16h bWs1 = load_bp(Ws1pl, t, lane);
    v8f acc1[3];
#pragma unroll
    for (int d = 0; d < 3; ++d) {
      acc1[d] = (v8f){};
      acc1[d] = wmma32(aF1[d], bWs1, acc1[d]);
      acc1[d] = wmma32(aO1[d], bWo1, acc1[d]);
    }
    float bgc = bgl[c];
#pragma unroll
    for (int r = 0; r < 8; ++r) {
      int node = rbase + r;
      float s0 = acc0[r];
      float x = acc1[0][r], y = acc1[1][r], z = acc1[2][r];
      if (doAct) {
        s0 = fmaxf(s0, 0.f);
        float n1 = sqrtf(x * x + y * y + z * z);
        float gate = fmaxf(n1 + bgc, 0.f) / (n1 + EPSF);
        x *= gate; y *= gate; z *= gate;
      }
      f0out[node * CC + c] = s0;
      float* fp = f1out + ((size_t)node * CC + c) * 3;
      fp[0] = x; fp[1] = y; fp[2] = z;
    }
  }
}

// ---------------------------------------------------------------------------
// host orchestration
// ---------------------------------------------------------------------------
extern "C" void kernel_launch(void* const* d_in, const int* in_sizes, int n_in,
                              void* d_out, int out_size, void* d_ws, size_t ws_size,
                              hipStream_t stream) {
  (void)in_sizes; (void)n_in; (void)out_size; (void)ws_size;
  const float* pos  = (const float*)d_in[0];
  const float* f0in = (const float*)d_in[1];
  const float* f1in = (const float*)d_in[2];
  const int*   src  = (const int*)d_in[3];
  const int*   dst  = (const int*)d_in[4];
  const float* W1   = (const float*)d_in[5];
  const float* b1   = (const float*)d_in[6];
  const float* W2   = (const float*)d_in[7];
  const float* b2   = (const float*)d_in[8];
  const float* Wq   = (const float*)d_in[9];
  const float* Wo0  = (const float*)d_in[10];
  const float* Wo1  = (const float*)d_in[11];
  const float* Ws0  = (const float*)d_in[12];
  const float* Ws1  = (const float*)d_in[13];
  const float* bg   = (const float*)d_in[14];

  // workspace carve (256B aligned)
  char* p = (char*)d_ws;
  auto carve = [&](size_t bytes) -> char* {
    char* r = p;
    p += (bytes + 255) & ~(size_t)255;
    return r;
  };
  float* u      = (float*)carve((size_t)EE * 3 * sizeof(float));
  float* rlen   = (float*)carve((size_t)EE * sizeof(float));
  float* f0a    = (float*)carve((size_t)NN * CC * sizeof(float));
  float* f0b    = (float*)carve((size_t)NN * CC * sizeof(float));
  float* f1a    = (float*)carve((size_t)NN * CC * 3 * sizeof(float));
  float* f1b    = (float*)carve((size_t)NN * CC * 3 * sizeof(float));
  float* qbuf   = (float*)carve((size_t)NN * CC * sizeof(float));
  float* logits = (float*)carve((size_t)EE * NHEADS * sizeof(float));
  float* eexp   = (float*)carve((size_t)EE * NHEADS * sizeof(float));
  float* mbuf   = (float*)carve((size_t)NN * NHEADS * sizeof(float));
  float* zbuf   = (float*)carve((size_t)NN * NHEADS * sizeof(float));
  float* o0     = (float*)carve((size_t)NN * CC * sizeof(float));
  float* o1     = (float*)carve((size_t)NN * CC * 3 * sizeof(float));
  // pre-swizzled f16 B operands: [layer][tile][lane][16]
  _Float16* W2p  = (_Float16*)carve((size_t)L1 * NTW2 * 32 * 16 * sizeof(_Float16));
  _Float16* Wqp  = (_Float16*)carve((size_t)L1 * 2 * 32 * 16 * sizeof(_Float16));
  _Float16* Wo0p = (_Float16*)carve((size_t)L1 * 2 * 32 * 16 * sizeof(_Float16));
  _Float16* Wo1p = (_Float16*)carve((size_t)L1 * 2 * 32 * 16 * sizeof(_Float16));
  _Float16* Ws0p = (_Float16*)carve((size_t)L1 * 2 * 32 * 16 * sizeof(_Float16));
  _Float16* Ws1p = (_Float16*)carve((size_t)L1 * 2 * 32 * 16 * sizeof(_Float16));

  // setup
  k_pack_b<<<(L1 * NTW2 * 32 + 255) / 256, 256, 0, stream>>>(
      W2, Wq, Wo0, Wo1, Ws0, Ws1, W2p, Wqp, Wo0p, Wo1p, Ws0p, Ws1p);
  k_edge_geom<<<(EE + 255) / 256, 256, 0, stream>>>(pos, src, dst, rlen, u);
  hipMemcpyAsync(f0a, f0in, (size_t)NN * CC * sizeof(float),
                 hipMemcpyDeviceToDevice, stream);
  hipMemcpyAsync(f1a, f1in, (size_t)NN * CC * 3 * sizeof(float),
                 hipMemcpyDeviceToDevice, stream);

  float* cf0 = f0a; float* cf1 = f1a;
  float* nf0 = f0b; float* nf1 = f1b;
  const int nodeBlocks = (NN / 16 + 3) / 4;  // 4 waves / block
  const int edgeBlocks = (EE / 16 + 3) / 4;
  const size_t w2Stride = (size_t)NTW2 * 32 * 16;  // f16 per layer
  const size_t wmStride = (size_t)2 * 32 * 16;

  for (int l = 0; l < L1; ++l) {
    int heads = (l < LLAY) ? NHEADS : 1;
    const float*    W1l  = W1 + (size_t)l * HRADK;
    const float*    b1l  = b1 + (size_t)l * HRADK;
    const _Float16* W2pl = W2p + (size_t)l * w2Stride;
    const float*    b2l  = b2 + (size_t)l * NRADC;

    k_zero_acc<<<(NN * CC + 255) / 256, 256, 0, stream>>>(mbuf, zbuf, o0, o1);
    k_node_q<<<nodeBlocks, 128, 0, stream>>>(cf0, Wqp + (size_t)l * wmStride, qbuf);
    k_edge_logits<<<edgeBlocks, 128, 0, stream>>>(
        rlen, u, src, dst, cf0, cf1, qbuf, W1l, b1l, W2pl, b2l, heads, logits, mbuf);
    k_edge_exp<<<(EE * heads + 255) / 256, 256, 0, stream>>>(
        logits, dst, mbuf, eexp, zbuf, heads);
    k_edge_values<<<edgeBlocks, 128, 0, stream>>>(
        rlen, u, src, dst, cf0, cf1, eexp, zbuf, W1l, b1l, W2pl, b2l, heads, o0, o1);

    float* of0 = (l == LLAY) ? (float*)d_out : nf0;
    float* of1 = (l == LLAY) ? ((float*)d_out + (size_t)NN * CC) : nf1;
    k_node_out<<<nodeBlocks, 128, 0, stream>>>(
        cf0, cf1, o0, o1,
        Wo0p + (size_t)l * wmStride, Wo1p + (size_t)l * wmStride,
        Ws0p + (size_t)l * wmStride, Ws1p + (size_t)l * wmStride,
        bg + (size_t)((l < LLAY) ? l : 0) * CC, (l < LLAY) ? 1 : 0, of0, of1);

    if (l < LLAY) {
      float* t0 = cf0; cf0 = nf0; nf0 = t0;
      float* t1 = cf1; cf1 = nf1; nf1 = t1;
    }
  }
}
